// DRNA_Block_11269994184841
// MI455X (gfx1250) — matrix operand order repaired and checked by reading, not compiled, not run on hardware
//
#include <hip/hip_runtime.h>
#include <math.h>

// ---------------------------------------------------------------------------
// Types for CDNA5 WMMA (wave32). 16x16x32 bf16 -> f32 accumulate.
// ---------------------------------------------------------------------------
typedef __attribute__((ext_vector_type(16))) __bf16 v16bf;
typedef __attribute__((ext_vector_type(8)))  __bf16 v8bf;
typedef __attribute__((ext_vector_type(8)))  float  v8f;

#define WMMA_BF16(a, b, c) \
    __builtin_amdgcn_wmma_f32_16x16x32_bf16(false, (a), false, (b), (short)0, (c), false, false)

static __device__ inline v16bf cat8(v8bf lo, v8bf hi) {
    return __builtin_shufflevector(lo, hi, 0,1,2,3,4,5,6,7,8,9,10,11,12,13,14,15);
}

// Raw 32-bit LDS offset of a __shared__ object (generic -> AS(3) -> int).
typedef __attribute__((address_space(3))) void lds_void_t;
static __device__ inline unsigned lds_off(const void* p) {
    return (unsigned)(unsigned long long)(lds_void_t*)p;
}

// Async DMA: global -> LDS, 16 bytes per lane. IOFFSET applies to BOTH the
// LDS and the global address (ISA 10.x async pseudocode), so one address
// setup serves multiple contiguous chunks. Tracked by ASYNCcnt.
static __device__ inline void async_b128(unsigned laddr, const void* gaddr) {
    unsigned long long ga = (unsigned long long)gaddr;
    asm volatile("global_load_async_to_lds_b128 %0, %1, off" :: "v"(laddr), "v"(ga) : "memory");
}
static __device__ inline void async_b128_off16(unsigned laddr, const void* gaddr) {
    unsigned long long ga = (unsigned long long)gaddr;
    asm volatile("global_load_async_to_lds_b128 %0, %1, off offset:16" :: "v"(laddr), "v"(ga) : "memory");
}
static __device__ inline void wait_async0() {
    asm volatile("s_wait_asynccnt 0" ::: "memory");
}

// ---------------------------------------------------------------------------
// Model constants
// ---------------------------------------------------------------------------
#define BB 2
#define SS 2048
#define DM 1024
#define NH 16
#define DH 64
#define DFF 4096
#define MM (BB * SS)   // 4096 rows

// ---------------------------------------------------------------------------
// f32 -> bf16 cast
// ---------------------------------------------------------------------------
__global__ __launch_bounds__(256) void cast_bf16_kernel(const float* __restrict__ src,
                                                        __bf16* __restrict__ dst, int n) {
    int i = blockIdx.x * 256 + threadIdx.x;
    if (i < n) dst[i] = (__bf16)src[i];
}

// ---------------------------------------------------------------------------
// C[M,N] (f32) = A[M,K] (bf16, row-major) * B[K,N] (bf16, row-major)
// 256 threads / 8 waves; tile 128x128, BK=32. Wave grid 4x2; each wave owns
// a 32x64 sub-tile = 2x4 wmma accumulators.
// Double-buffered LDS; A tile prefetched with async-load-to-LDS (DMA path,
// no VGPR round trip); B tile staged transposed via ds stores overlapped
// with compute. Two barriers per K-step.
// ---------------------------------------------------------------------------
__global__ __launch_bounds__(256) void gemm_bf16_kernel(const __bf16* __restrict__ A,
                                                        const __bf16* __restrict__ B,
                                                        float* __restrict__ C,
                                                        int M, int N, int K) {
    constexpr int BM = 128, BN = 128, BK = 32;
    constexpr int ASTR = BK + 8;   // 40
    constexpr int BSTR = BK + 8;   // B^T stride
    __shared__ __align__(32) __bf16 sA[2][BM * ASTR];
    __shared__ __align__(32) __bf16 sBT[2][BN * BSTR];

    const int tid  = threadIdx.x;
    const int lane = tid & 31;
    const int wave = tid >> 5;
    const int wm   = wave >> 1;    // 0..3
    const int wn   = wave & 1;     // 0..1
    const int half = lane >> 4;    // 0/1
    const int l16  = lane & 15;

    const int tileM = blockIdx.y * BM;
    const int tileN = blockIdx.x * BN;

    auto stage = [&](int buf, int k0) {
        // A tile 128x32: async DMA, two b128 per thread, contiguous in both
        // global and (padded) LDS layouts.
        {
            int idx = tid * 16;            // 0..4095
            int row = idx >> 5;
            int kk  = idx & 31;            // 0 or 16
            const __bf16* g = A + (size_t)(tileM + row) * K + k0 + kk;
            unsigned l0 = lds_off(&sA[buf][row * ASTR + kk]);
            async_b128(l0, g);
            async_b128_off16(l0, g);
        }
        // B tile 32x128 -> transposed sBT[n][k] (ds stores)
        {
            int idx = tid * 16;
            int kr  = idx >> 7;            // 0..31
            int n0  = idx & 127;
            const __bf16* g = B + (size_t)(k0 + kr) * N + tileN + n0;
            v8bf p0 = *(const v8bf*)(g);
            v8bf p1 = *(const v8bf*)(g + 8);
#pragma unroll
            for (int e = 0; e < 8; ++e) sBT[buf][(n0 + e) * BSTR + kr]     = p0[e];
#pragma unroll
            for (int e = 0; e < 8; ++e) sBT[buf][(n0 + 8 + e) * BSTR + kr] = p1[e];
        }
    };

    const v8f vzero = {0.f, 0.f, 0.f, 0.f, 0.f, 0.f, 0.f, 0.f};
    v8f acc[2][4];
#pragma unroll
    for (int i = 0; i < 2; ++i)
#pragma unroll
        for (int j = 0; j < 4; ++j) acc[i][j] = vzero;

    const int NK = K / BK;
    stage(0, 0);

    for (int kt = 0; kt < NK; ++kt) {
        const int cur = kt & 1;
        wait_async0();          // this wave's async writes into sA[cur] done
        __syncthreads();        // everyone's staging (async + ds) visible
        if (kt + 1 < NK) stage(cur ^ 1, (kt + 1) * BK);   // prefetch overlaps compute

        // A fragments (16x32): lane row = l%16; runs half*8 and 16+half*8
        v16bf af[2];
#pragma unroll
        for (int i = 0; i < 2; ++i) {
            const __bf16* p = &sA[cur][(wm * 32 + i * 16 + l16) * ASTR];
            v8bf lo = *(const v8bf*)(p + half * 8);
            v8bf hi = *(const v8bf*)(p + 16 + half * 8);
            af[i] = cat8(lo, hi);
        }
        // B fragments (32x16): lane col = l%16; runs half*8 and 16+half*8
        v16bf bfrag[4];
#pragma unroll
        for (int j = 0; j < 4; ++j) {
            const __bf16* p = &sBT[cur][(wn * 64 + j * 16 + l16) * BSTR + half * 8];
            v8bf lo = *(const v8bf*)(p);
            v8bf hi = *(const v8bf*)(p + 16);
            bfrag[j] = cat8(lo, hi);
        }
#pragma unroll
        for (int i = 0; i < 2; ++i)
#pragma unroll
            for (int j = 0; j < 4; ++j)
                acc[i][j] = WMMA_BF16(af[i], bfrag[j], acc[i][j]);
        __syncthreads();        // reads of sA[cur]/sBT[cur] done before next overwrite
    }

    // C/D layout: VGPR r -> M = r + 8*half ; lane -> N = l%16
#pragma unroll
    for (int i = 0; i < 2; ++i)
#pragma unroll
        for (int j = 0; j < 4; ++j) {
            int n = tileN + wn * 64 + j * 16 + l16;
#pragma unroll
            for (int r = 0; r < 8; ++r) {
                int m = tileM + wm * 32 + i * 16 + r + 8 * half;
                C[(size_t)m * N + n] = acc[i][j][r];
            }
        }
}

// ---------------------------------------------------------------------------
// QKV bias + RoPE + split into [B,H,S,DH] bf16 (q,k rotated; v plain)
// ---------------------------------------------------------------------------
__global__ __launch_bounds__(256) void qkv_rope_split_kernel(const float* __restrict__ qkv,
                                                             const float* __restrict__ bqkv,
                                                             const float* __restrict__ cosb,
                                                             const float* __restrict__ sinb,
                                                             __bf16* __restrict__ Qo,
                                                             __bf16* __restrict__ Ko,
                                                             __bf16* __restrict__ Vo) {
    int idx = blockIdx.x * 256 + threadIdx.x;        // over B*S*H*DH
    int d  = idx & 63;
    int h  = (idx >> 6) & (NH - 1);
    int bs = idx >> 10;
    int s  = bs & (SS - 1);
    int b  = bs >> 11;                               // SS = 2048 = 1<<11
    size_t row = (size_t)b * SS + s;
    const float* base = qkv + row * (3 * DM);

    int dp = (d < 32) ? d + 32 : d - 32;
    int cq  = h * DH + d,  cqp = h * DH + dp;
    int ck  = DM + cq,     ckp = DM + cqp;
    int cv  = 2 * DM + cq;

    float qv  = base[cq]  + bqkv[cq];
    float qpv = base[cqp] + bqkv[cqp];
    float kv  = base[ck]  + bqkv[ck];
    float kpv = base[ckp] + bqkv[ckp];
    float vv  = base[cv]  + bqkv[cv];

    float c = cosb[row * DH + d];
    float sn = sinb[row * DH + d];
    float rotq = (d < 32) ? -qpv : qpv;
    float rotk = (d < 32) ? -kpv : kpv;

    size_t o = (((size_t)b * NH + h) * SS + s) * DH + d;
    Qo[o] = (__bf16)(qv * c + rotq * sn);
    Ko[o] = (__bf16)(kv * c + rotk * sn);
    Vo[o] = (__bf16)vv;
}

// ---------------------------------------------------------------------------
// Flash attention. Block = 256 threads = 8 waves, one (b,h,128-query) slab.
// Each wave owns a 16-query tile; K/V staged in LDS in 32-key tiles
// (K via async DMA, V transposed via ds stores).
// Output written as [B, S, H*DH] bf16.
// ---------------------------------------------------------------------------
__global__ __launch_bounds__(256) void flash_attn_kernel(const __bf16* __restrict__ Q,
                                                         const __bf16* __restrict__ K,
                                                         const __bf16* __restrict__ V,
                                                         __bf16* __restrict__ O) {
    constexpr int KSTR = DH + 8;    // 72
    constexpr int VSTR = 32 + 8;    // 40 (V^T rows of 32 keys)
    constexpr int PSTR = 32 + 8;    // per-wave P tile stride
    __shared__ __align__(32) __bf16 sK[32 * KSTR];
    __shared__ __align__(32) __bf16 sVT[DH * VSTR];
    __shared__ __align__(32) __bf16 sP[8 * 16 * PSTR];

    const int tid  = threadIdx.x;
    const int lane = tid & 31;
    const int wave = tid >> 5;
    const int half = lane >> 4;
    const int l16  = lane & 15;

    const int nqb = SS / 128;                  // 16
    int b   = blockIdx.x / (NH * nqb);
    int rem = blockIdx.x % (NH * nqb);
    int h   = rem / nqb;
    int qb  = rem % nqb;
    int q0  = qb * 128 + wave * 16;

    const __bf16* qp = Q + (((size_t)b * NH + h) * SS) * DH;
    const __bf16* kp = K + (((size_t)b * NH + h) * SS) * DH;
    const __bf16* vp = V + (((size_t)b * NH + h) * SS) * DH;

    // Q fragments: two K-steps over d (0..31, 32..63)
    v16bf qf[2];
#pragma unroll
    for (int s = 0; s < 2; ++s) {
        const __bf16* base = qp + (size_t)(q0 + l16) * DH + s * 32;
        v8bf lo = *(const v8bf*)(base + half * 8);
        v8bf hi = *(const v8bf*)(base + 16 + half * 8);
        qf[s] = cat8(lo, hi);
    }

    const v8f vzero = {0.f, 0.f, 0.f, 0.f, 0.f, 0.f, 0.f, 0.f};
    v8f o_acc[4];
#pragma unroll
    for (int c = 0; c < 4; ++c) o_acc[c] = vzero;
    float mrow[8], lrow[8];
#pragma unroll
    for (int j = 0; j < 8; ++j) { mrow[j] = -INFINITY; lrow[j] = 0.f; }

    const float scale = 0.125f;   // 1/sqrt(64)

    for (int kb = 0; kb < SS; kb += 32) {
        __syncthreads();                         // previous tiles fully consumed
        // stage K tile [key][d] via async DMA (one b128 per thread)
        {
            int idx = tid * 8;                   // 0..2047
            int key = idx >> 6, d0 = idx & 63;
            unsigned l0 = lds_off(&sK[key * KSTR + d0]);
            async_b128(l0, kp + (size_t)(kb + key) * DH + d0);
        }
        // stage V transposed -> sVT[d][key]
        {
            int idx = tid * 8;
            int key = idx >> 6, d0 = idx & 63;
            v8bf p = *(const v8bf*)(vp + (size_t)(kb + key) * DH + d0);
#pragma unroll
            for (int e = 0; e < 8; ++e) sVT[(d0 + e) * VSTR + key] = p[e];
        }
        wait_async0();
        __syncthreads();

        // scores: S(16x32) = Q(16x64) @ K^T(64x32) -> 2 key-subtiles, 2 chained K-steps
        v8f sc[2];
        sc[0] = vzero; sc[1] = vzero;
#pragma unroll
        for (int t = 0; t < 2; ++t)
#pragma unroll
            for (int s = 0; s < 2; ++s) {
                const __bf16* p = sK + (t * 16 + l16) * KSTR + s * 32 + half * 8;
                v8bf lo = *(const v8bf*)(p);
                v8bf hi = *(const v8bf*)(p + 16);
                sc[t] = WMMA_BF16(qf[s], cat8(lo, hi), sc[t]);
            }

        // online softmax; rows live in 16-lane groups of the C layout
        float corr[8];
#pragma unroll
        for (int j = 0; j < 8; ++j) {
            float s0 = sc[0][j] * scale;
            float s1 = sc[1][j] * scale;
            float mx = fmaxf(s0, s1);
#pragma unroll
            for (int dmask = 1; dmask < 16; dmask <<= 1)
                mx = fmaxf(mx, __shfl_xor(mx, dmask, 32));
            float mnew = fmaxf(mrow[j], mx);
            float p0 = __expf(s0 - mnew);
            float p1 = __expf(s1 - mnew);
            float cf = __expf(mrow[j] - mnew);
            float rs = p0 + p1;
#pragma unroll
            for (int dmask = 1; dmask < 16; dmask <<= 1)
                rs += __shfl_xor(rs, dmask, 32);
            lrow[j] = lrow[j] * cf + rs;
            mrow[j] = mnew;
            corr[j] = cf;
            // C-layout -> LDS (per-wave region): row = j + 8*half, col = t*16 + l16
            __bf16* pw = sP + (wave * 16 + j + 8 * half) * PSTR;
            pw[l16]      = (__bf16)p0;
            pw[16 + l16] = (__bf16)p1;
        }
        // rescale running O
#pragma unroll
        for (int c = 0; c < 4; ++c)
#pragma unroll
            for (int j = 0; j < 8; ++j) o_acc[c][j] *= corr[j];

        // P fragment (A-layout, 16x32 over keys) from per-wave LDS
        v16bf pf;
        {
            const __bf16* pb = sP + (wave * 16 + l16) * PSTR;
            v8bf lo = *(const v8bf*)(pb + half * 8);
            v8bf hi = *(const v8bf*)(pb + 16 + half * 8);
            pf = cat8(lo, hi);
        }
        // O += P(16x32) @ V(32x64): 4 d-chunks
#pragma unroll
        for (int c = 0; c < 4; ++c) {
            const __bf16* vb = sVT + (c * 16 + l16) * VSTR + half * 8;
            v8bf lo = *(const v8bf*)(vb);
            v8bf hi = *(const v8bf*)(vb + 16);
            o_acc[c] = WMMA_BF16(pf, cat8(lo, hi), o_acc[c]);
        }
    }

    // epilogue: O /= l, scatter to [b, s, h*64 + d]
#pragma unroll
    for (int c = 0; c < 4; ++c) {
        int d = c * 16 + l16;
#pragma unroll
        for (int j = 0; j < 8; ++j) {
            int row = q0 + j + 8 * half;
            float val = o_acc[c][j] / lrow[j];
            O[((size_t)(b * SS + row) * NH + h) * DH + d] = (__bf16)val;
        }
    }
}

// ---------------------------------------------------------------------------
// y = LayerNorm(resid + c + bias) * g + beta ; writes f32 (+ optional bf16)
// one block per row, 256 threads, D = 1024 (4 elems/thread)
// ---------------------------------------------------------------------------
__global__ __launch_bounds__(256) void bias_residual_ln_kernel(const float* __restrict__ resid,
                                                               const float* __restrict__ c,
                                                               const float* __restrict__ bias,
                                                               const float* __restrict__ g,
                                                               const float* __restrict__ beta,
                                                               float* __restrict__ out_f32,
                                                               __bf16* __restrict__ out_bf) {
    __shared__ float red1[8];
    __shared__ float red2[8];
    const int row  = blockIdx.x;
    const int tid  = threadIdx.x;
    const int lane = tid & 31;
    const int wave = tid >> 5;
    const float* cr = c + (size_t)row * DM;
    const float* rr = resid + (size_t)row * DM;

    float v[4];
    float sum = 0.f;
#pragma unroll
    for (int i = 0; i < 4; ++i) {
        int col = tid + i * 256;
        v[i] = rr[col] + cr[col] + bias[col];
        sum += v[i];
    }
#pragma unroll
    for (int dmask = 16; dmask >= 1; dmask >>= 1) sum += __shfl_xor(sum, dmask, 32);
    if (lane == 0) red1[wave] = sum;
    __syncthreads();
    float tot = 0.f;
#pragma unroll
    for (int w = 0; w < 8; ++w) tot += red1[w];
    float mu = tot * (1.0f / DM);

    float vs = 0.f;
#pragma unroll
    for (int i = 0; i < 4; ++i) { float dd = v[i] - mu; vs += dd * dd; }
#pragma unroll
    for (int dmask = 16; dmask >= 1; dmask >>= 1) vs += __shfl_xor(vs, dmask, 32);
    if (lane == 0) red2[wave] = vs;
    __syncthreads();
    float tot2 = 0.f;
#pragma unroll
    for (int w = 0; w < 8; ++w) tot2 += red2[w];
    float rstd = rsqrtf(tot2 * (1.0f / DM) + 1e-5f);

#pragma unroll
    for (int i = 0; i < 4; ++i) {
        int col = tid + i * 256;
        float y = (v[i] - mu) * rstd * g[col] + beta[col];
        out_f32[(size_t)row * DM + col] = y;
        if (out_bf) out_bf[(size_t)row * DM + col] = (__bf16)y;
    }
}

// ---------------------------------------------------------------------------
// h = gelu(c + bias) -> bf16  (exact erf gelu)
// ---------------------------------------------------------------------------
__global__ __launch_bounds__(256) void bias_gelu_kernel(const float* __restrict__ c,
                                                        const float* __restrict__ bias,
                                                        __bf16* __restrict__ out, int N) {
    int i = blockIdx.x * 256 + threadIdx.x;
    int n = i % N;
    float h = c[i] + bias[n];
    float y = 0.5f * h * (1.0f + erff(h * 0.70710678118654752f));
    out[i] = (__bf16)y;
}

// ---------------------------------------------------------------------------
// Host-side orchestration
// ---------------------------------------------------------------------------
extern "C" void kernel_launch(void* const* d_in, const int* in_sizes, int n_in,
                              void* d_out, int out_size, void* d_ws, size_t ws_size,
                              hipStream_t stream) {
    (void)in_sizes; (void)n_in; (void)out_size; (void)ws_size;
    const float* x     = (const float*)d_in[0];
    const float* cosb  = (const float*)d_in[1];
    const float* sinb  = (const float*)d_in[2];
    const float* w_qkv = (const float*)d_in[3];
    const float* b_qkv = (const float*)d_in[4];
    const float* w_out = (const float*)d_in[5];
    const float* b_out = (const float*)d_in[6];
    const float* ln1_g = (const float*)d_in[7];
    const float* ln1_b = (const float*)d_in[8];
    const float* w1    = (const float*)d_in[9];
    const float* b1    = (const float*)d_in[10];
    const float* w2    = (const float*)d_in[11];
    const float* b2    = (const float*)d_in[12];
    const float* ln2_g = (const float*)d_in[13];
    const float* ln2_b = (const float*)d_in[14];
    float* out = (float*)d_out;

    size_t off = 0;
    auto carve = [&](size_t bytes) -> char* {
        char* p = (char*)d_ws + off;
        off += (bytes + 255) & ~(size_t)255;
        return p;
    };
    __bf16* wqkv_bf  = (__bf16*)carve((size_t)DM * 3 * DM * 2);
    __bf16* wout_bf  = (__bf16*)carve((size_t)DM * DM * 2);
    __bf16* w1_bf    = (__bf16*)carve((size_t)DM * DFF * 2);
    __bf16* w2_bf    = (__bf16*)carve((size_t)DFF * DM * 2);
    __bf16* x_bf     = (__bf16*)carve((size_t)MM * DM * 2);
    float*  qkv_f32  = (float*) carve((size_t)MM * 3 * DM * 4);
    __bf16* q_bf     = (__bf16*)carve((size_t)MM * DM * 2);
    __bf16* k_bf     = (__bf16*)carve((size_t)MM * DM * 2);
    __bf16* v_bf     = (__bf16*)carve((size_t)MM * DM * 2);
    __bf16* aout_bf  = (__bf16*)carve((size_t)MM * DM * 2);
    float*  proj_f32 = (float*) carve((size_t)MM * DM * 4);
    float*  ln1_f32  = (float*) carve((size_t)MM * DM * 4);
    __bf16* ln1_bf   = (__bf16*)carve((size_t)MM * DM * 2);
    float*  ffn1_f32 = (float*) carve((size_t)MM * DFF * 4);
    __bf16* h_bf     = (__bf16*)carve((size_t)MM * DFF * 2);
    float*  ffn2_f32 = (float*) carve((size_t)MM * DM * 4);

    // 1) bf16 casts
    cast_bf16_kernel<<<(MM * DM) / 256, 256, 0, stream>>>(x, x_bf, MM * DM);
    cast_bf16_kernel<<<(DM * 3 * DM) / 256, 256, 0, stream>>>(w_qkv, wqkv_bf, DM * 3 * DM);
    cast_bf16_kernel<<<(DM * DM) / 256, 256, 0, stream>>>(w_out, wout_bf, DM * DM);
    cast_bf16_kernel<<<(DM * DFF) / 256, 256, 0, stream>>>(w1, w1_bf, DM * DFF);
    cast_bf16_kernel<<<(DFF * DM) / 256, 256, 0, stream>>>(w2, w2_bf, DFF * DM);

    // 2) QKV GEMM: [4096,1024] @ [1024,3072]
    gemm_bf16_kernel<<<dim3(3 * DM / 128, MM / 128), 256, 0, stream>>>(
        x_bf, wqkv_bf, qkv_f32, MM, 3 * DM, DM);

    // 3) bias + RoPE + split
    qkv_rope_split_kernel<<<(MM * DM) / 256, 256, 0, stream>>>(
        qkv_f32, b_qkv, cosb, sinb, q_bf, k_bf, v_bf);

    // 4) flash attention
    flash_attn_kernel<<<BB * NH * (SS / 128), 256, 0, stream>>>(q_bf, k_bf, v_bf, aout_bf);

    // 5) out projection: [4096,1024] @ [1024,1024]
    gemm_bf16_kernel<<<dim3(DM / 128, MM / 128), 256, 0, stream>>>(
        aout_bf, wout_bf, proj_f32, MM, DM, DM);

    // 6) x1 = LN(x + proj + b_out)
    bias_residual_ln_kernel<<<MM, 256, 0, stream>>>(
        x, proj_f32, b_out, ln1_g, ln1_b, ln1_f32, ln1_bf);

    // 7) FFN up: [4096,1024] @ [1024,4096]
    gemm_bf16_kernel<<<dim3(DFF / 128, MM / 128), 256, 0, stream>>>(
        ln1_bf, w1_bf, ffn1_f32, MM, DFF, DM);

    // 8) gelu(. + b1) -> bf16
    bias_gelu_kernel<<<(MM * DFF) / 256, 256, 0, stream>>>(ffn1_f32, b1, h_bf, DFF);

    // 9) FFN down: [4096,4096] @ [4096,1024]
    gemm_bf16_kernel<<<dim3(DM / 128, MM / 128), 256, 0, stream>>>(
        h_bf, w2_bf, ffn2_f32, MM, DM, DFF);

    // 10) out = LN(x1 + ffn2 + b2)
    bias_residual_ln_kernel<<<MM, 256, 0, stream>>>(
        ln1_f32, ffn2_f32, b2, ln2_g, ln2_b, out, nullptr);
}
